// contrastiveLoss_80711025426432
// MI455X (gfx1250) — matrix operand order repaired and checked by reading, not compiled, not run on hardware
//
#include <hip/hip_runtime.h>
#include <hip/hip_bf16.h>

typedef __attribute__((ext_vector_type(16))) _Float16 v16h;
typedef __attribute__((ext_vector_type(8)))  _Float16 v8h;
typedef __attribute__((ext_vector_type(8)))  float    v8f;

#define CN      8192
#define CD      64
#define C_LAM   0.5f
#define C_INVT  1.25f      // 1 / tau, tau = 0.8
#define C_EPS   1e-8f

// ---------------------------------------------------------------------------
// Kernel 0: zero the 4 accumulator arrays (contiguous, 4*CN floats)
// ---------------------------------------------------------------------------
__global__ void k_zero(float* __restrict__ p, int n) {
    int i = blockIdx.x * blockDim.x + threadIdx.x;
    if (i < n) p[i] = 0.0f;
}

// ---------------------------------------------------------------------------
// Kernel 1: row-normalize z_mp and z_sc into f16 matrices (one wave per row)
// ---------------------------------------------------------------------------
__global__ void k_normalize(const float* __restrict__ zmp,
                            const float* __restrict__ zsc,
                            _Float16* __restrict__ Zm,
                            _Float16* __restrict__ Zs) {
    int wave = (blockIdx.x * blockDim.x + threadIdx.x) >> 5;
    int lane = threadIdx.x & 31;
    if (wave >= 2 * CN) return;
    const float* src = (wave < CN) ? zmp : zsc;
    _Float16*    dst = (wave < CN) ? Zm  : Zs;
    int row = (wave < CN) ? wave : wave - CN;

    float2 v = ((const float2*)(src + (size_t)row * CD))[lane];
    float ss = v.x * v.x + v.y * v.y;
#pragma unroll
    for (int m = 16; m >= 1; m >>= 1) ss += __shfl_xor(ss, m, 32);
    float inv = rsqrtf(ss);
    dst[(size_t)row * CD + 2 * lane]     = (_Float16)(v.x * inv);
    dst[(size_t)row * CD + 2 * lane + 1] = (_Float16)(v.y * inv);
}

// ---------------------------------------------------------------------------
// Fragment loaders (wave32 CDNA5 WMMA 16x16x32 f16 layouts)
// A 16x32: lane r (r=lane&15, hi=lane>>4) holds row r,
//          K = {hi*8 .. hi*8+7} in v[0..7], K = {16+hi*8 ..} in v[8..15]
// B 32x16 (we feed B^T row-major): lane r holds column r,
//          K = {hi*8 .. hi*8+7} in v[0..7], K = {16+hi*8 ..} in v[8..15]
// ---------------------------------------------------------------------------
__device__ inline v16h load_afrag(const _Float16* __restrict__ Z,
                                  int I0, int k0, int r, int hi) {
    const _Float16* base = Z + (size_t)(I0 + r) * CD + k0 + hi * 8;
    v8h lo = *(const v8h*)(base);
    v8h hh = *(const v8h*)(base + 16);
    v16h a;
#pragma unroll
    for (int i = 0; i < 8; ++i) { a[i] = lo[i]; a[8 + i] = hh[i]; }
    return a;
}

__device__ inline v16h load_bfrag(const _Float16* __restrict__ Z,
                                  int J0, int k0, int r, int hi) {
    const _Float16* base = Z + (size_t)(J0 + r) * CD + k0 + hi * 8;
    v8h lo = *(const v8h*)(base);
    v8h hh = *(const v8h*)(base + 16);
    v16h b;
#pragma unroll
    for (int i = 0; i < 8; ++i) { b[i] = lo[i]; b[8 + i] = hh[i]; }
    return b;
}

// ---------------------------------------------------------------------------
// Reduce one 16x16 tile of S = exp(dot * invTau) into the 4 accumulators.
// C/D layout: VGPR v, lanes 0-15 -> (M=v, N=lane), lanes 16-31 -> (M=v+8).
// Element S[I0+m][J0+n] contributes:
//   rowsum[I0+m]    += S
//   possum_row[I0+m]+= S * pos[I0+m][J0+n]
//   colsum[J0+n]    += S
//   possum_col[J0+n]+= S * pos[J0+n][I0+m]   (transposed pos!)
// ---------------------------------------------------------------------------
__device__ inline void reduce_tile(const v8f& c, int I0, int J0,
                                   const int* __restrict__ pos,
                                   float* __restrict__ rowsum,
                                   float* __restrict__ prow,
                                   float* __restrict__ colsum,
                                   float* __restrict__ pcol,
                                   int r, int hi) {
    float e[8];
#pragma unroll
    for (int v = 0; v < 8; ++v) e[v] = __expf(c[v] * C_INVT);

    // ---- column-side sums: column n = J0 + r, rows m = I0 + hi*8 + v ----
    const int* posT = pos + (size_t)(J0 + r) * CN + I0 + hi * 8; // contiguous 8
    int4 p0 = ((const int4*)posT)[0];
    int4 p1 = ((const int4*)posT)[1];
    int pt[8] = {p0.x, p0.y, p0.z, p0.w, p1.x, p1.y, p1.z, p1.w};
    float cs = 0.f, pcs = 0.f;
#pragma unroll
    for (int v = 0; v < 8; ++v) { cs += e[v]; pcs += e[v] * (float)pt[v]; }
    cs  += __shfl_xor(cs, 16, 32);
    pcs += __shfl_xor(pcs, 16, 32);
    if (hi == 0) {
        atomicAdd(colsum + J0 + r, cs);
        atomicAdd(pcol   + J0 + r, pcs);
    }

    // ---- row-side sums: row m = I0 + hi*8 + v, weight pos[m][J0+r] ----
#pragma unroll
    for (int v = 0; v < 8; ++v) {
        float rs  = e[v];
        float prs = e[v] * (float)pos[(size_t)(I0 + hi * 8 + v) * CN + J0 + r];
#pragma unroll
        for (int m = 1; m <= 8; m <<= 1) {   // reduce over 16-lane half
            rs  += __shfl_xor(rs,  m, 32);
            prs += __shfl_xor(prs, m, 32);
        }
        if (r == 0) {
            atomicAdd(rowsum + I0 + hi * 8 + v, rs);
            atomicAdd(prow   + I0 + hi * 8 + v, prs);
        }
    }
}

// ---------------------------------------------------------------------------
// Kernel 2: fused similarity + reductions. One wave per tile-pair
// (ti,tj)/(tj,ti), ti <= tj, so each pos tile is touched by exactly one wave
// in both orientations (single HBM read per cacheline).
// ---------------------------------------------------------------------------
__global__ void __launch_bounds__(256)
k_main(const _Float16* __restrict__ Zm, const _Float16* __restrict__ Zs,
       const int* __restrict__ pos,
       float* __restrict__ rowsum, float* __restrict__ prow,
       float* __restrict__ colsum, float* __restrict__ pcol) {
    int w    = threadIdx.x >> 5;
    int lane = threadIdx.x & 31;
    int r = lane & 15, hi = lane >> 4;

    int ti = blockIdx.x;
    int tj = blockIdx.y * 8 + w;
    if (tj < ti) return;                 // wave-uniform: EXEC stays all-ones
    int I0 = ti * 16, J0 = tj * 16;
    bool off_diag = (ti != tj);          // wave-uniform

    v8f c1 = {0.f, 0.f, 0.f, 0.f, 0.f, 0.f, 0.f, 0.f};
    v8f c2 = {0.f, 0.f, 0.f, 0.f, 0.f, 0.f, 0.f, 0.f};
#pragma unroll
    for (int k0 = 0; k0 < CD; k0 += 32) {
        v16h am = load_afrag(Zm, I0, k0, r, hi);
        v16h bs = load_bfrag(Zs, J0, k0, r, hi);
        c1 = __builtin_amdgcn_wmma_f32_16x16x32_f16(
                 false, am, false, bs, (short)0, c1, false, false);
        if (off_diag) {
            v16h am2 = load_afrag(Zm, J0, k0, r, hi);
            v16h bs2 = load_bfrag(Zs, I0, k0, r, hi);
            c2 = __builtin_amdgcn_wmma_f32_16x16x32_f16(
                     false, am2, false, bs2, (short)0, c2, false, false);
        }
    }
    reduce_tile(c1, I0, J0, pos, rowsum, prow, colsum, pcol, r, hi);
    if (off_diag)
        reduce_tile(c2, J0, I0, pos, rowsum, prow, colsum, pcol, r, hi);
}

// ---------------------------------------------------------------------------
// Kernel 3: final scalar loss (single block)
// ---------------------------------------------------------------------------
__global__ void k_finalize(const float* __restrict__ rowsum,
                           const float* __restrict__ prow,
                           const float* __restrict__ colsum,
                           const float* __restrict__ pcol,
                           float* __restrict__ out) {
    __shared__ float sm[256];
    float acc = 0.f;
    for (int i = threadIdx.x; i < CN; i += 256) {
        float t1 = __logf(prow[i] / (rowsum[i] + C_EPS));
        float t2 = __logf(pcol[i] / (colsum[i] + C_EPS));
        acc += C_LAM * t1 + (1.0f - C_LAM) * t2;
    }
    sm[threadIdx.x] = acc;
    __syncthreads();
    for (int s = 128; s > 0; s >>= 1) {
        if (threadIdx.x < s) sm[threadIdx.x] += sm[threadIdx.x + s];
        __syncthreads();
    }
    if (threadIdx.x == 0) out[0] = -sm[0] / (float)CN;
}

// ---------------------------------------------------------------------------
extern "C" void kernel_launch(void* const* d_in, const int* in_sizes, int n_in,
                              void* d_out, int out_size, void* d_ws, size_t ws_size,
                              hipStream_t stream) {
    const float* zmp = (const float*)d_in[0];
    const float* zsc = (const float*)d_in[1];
    const int*   pos = (const int*)d_in[2];

    char* ws = (char*)d_ws;
    _Float16* Zm = (_Float16*)ws;  ws += (size_t)CN * CD * sizeof(_Float16);
    _Float16* Zs = (_Float16*)ws;  ws += (size_t)CN * CD * sizeof(_Float16);
    float* rowsum = (float*)ws;    ws += (size_t)CN * sizeof(float);
    float* prow   = (float*)ws;    ws += (size_t)CN * sizeof(float);
    float* colsum = (float*)ws;    ws += (size_t)CN * sizeof(float);
    float* pcol   = (float*)ws;    ws += (size_t)CN * sizeof(float);

    // rowsum/prow/colsum/pcol are contiguous: zero them in one shot
    k_zero<<<dim3((4 * CN + 255) / 256), 256, 0, stream>>>(rowsum, 4 * CN);

    // 2*CN rows, 8 waves (rows) per 256-thread block
    k_normalize<<<dim3((2 * CN) / 8), 256, 0, stream>>>(zmp, zsc, Zm, Zs);

    // 512 x 512 tiles; 8 waves/block handle 8 consecutive tj per block
    k_main<<<dim3(CN / 16, (CN / 16) / 8), 256, 0, stream>>>(
        Zm, Zs, pos, rowsum, prow, colsum, pcol);

    k_finalize<<<1, 256, 0, stream>>>(rowsum, prow, colsum, pcol, (float*)d_out);
}
